// SelfAttention_54004918780709
// MI455X (gfx1250) — matrix operand order repaired and checked by reading, not compiled, not run on hardware
//
#include <hip/hip_runtime.h>
#include <hip/hip_bf16.h>
#include <math.h>

typedef _Float16 half_t;
typedef __attribute__((ext_vector_type(16))) half_t v16h;
typedef __attribute__((ext_vector_type(8)))  float  v8f;
typedef __attribute__((ext_vector_type(4)))  unsigned int u32x4;
typedef __attribute__((ext_vector_type(8)))  int i32x8;
typedef __attribute__((ext_vector_type(4)))  int i32x4;

#define IN_F  1024
#define HEADS 16
#define HDIM  64
#define BATCH 4
#define SEQ   2048
#define ROWS  (BATCH * SEQ)   // 8192

#if __has_builtin(__builtin_amdgcn_tensor_load_to_lds) && \
    __has_builtin(__builtin_amdgcn_s_wait_tensorcnt)
#define USE_TDM 1
#else
#define USE_TDM 0
#endif

// ---------------------------------------------------------------- WMMA helper
__device__ __forceinline__ v8f wmma32(v16h a, v16h b, v8f c) {
  // D = A(16x32 f16) * B(32x16 f16) + C(16x16 f32)
  return __builtin_amdgcn_wmma_f32_16x16x32_f16(false, a, false, b, (short)0, c,
                                                false, false);
}

union FragU { v16h v; int4 q[2]; };

// A-fragment (16x32 f16) from row-major LDS tile [m][k], per ISA layout:
// lanes 0-15 hold K 0-7 & 16-23, lanes 16-31 hold K 8-15 & 24-31.
__device__ __forceinline__ v16h ldsA(const half_t* base, int stride, int mbase,
                                     int k0, int lane) {
  const half_t* p = base + (mbase + (lane & 15)) * stride + k0 + ((lane >> 4) << 3);
  FragU u;
  u.q[0] = *(const int4*)(p);
  u.q[1] = *(const int4*)(p + 16);
  return u.v;
}

// B-fragment (32x16 f16) from LDS tile stored transposed [n][k]:
// lanes 0-15 hold K 0-7 & 16-23, lanes 16-31 hold K 8-15 & 24-31.
__device__ __forceinline__ v16h ldsB(const half_t* base, int stride, int nbase,
                                     int k0, int lane) {
  const half_t* p = base + (nbase + (lane & 15)) * stride + k0 + ((lane >> 4) << 3);
  FragU u;
  u.q[0] = *(const int4*)(p);
  u.q[1] = *(const int4*)(p + 16);
  return u.v;
}

#if USE_TDM
// ---------------------------------------------------------------- TDM helper
// 2D f16 tile DMA: global (row-major, row_stride elems) -> LDS, with LDS row
// padding generated by the D# pad fields (ISA 8.4): row of 2^(pi+1) DWORDs,
// then (pa+1) DWORDs of pad. Issue from ONE wave only; tracked by TENSORcnt.
__device__ __forceinline__ void tdm_load_2d(unsigned int lds_byte_off,
                                            const void* gsrc,
                                            unsigned int width_elems,
                                            unsigned int height_rows,
                                            unsigned int row_stride_elems,
                                            unsigned int pad_interval,
                                            unsigned int pad_amount) {
  unsigned long long ga = (unsigned long long)(uintptr_t)gsrc;
  u32x4 g0;
  g0[0] = 1u;                                      // count=1, user descriptor
  g0[1] = lds_byte_off;                            // lds_addr
  g0[2] = (unsigned int)(ga & 0xffffffffu);        // global_addr[31:0]
  g0[3] = (unsigned int)((ga >> 32) & 0x1ffffffu)  // global_addr[56:32]
          | (2u << 30);                            // type=2 (image)
  i32x8 g1;
  g1[0] = (int)((1u << 16)                         // data_size=1 -> 2 bytes
                | (1u << 20)                       // pad_enable
                | (pad_interval << 22) | (pad_amount << 25));
  g1[1] = (int)((width_elems & 0xffffu) << 16);    // tensor_dim0[15:0]
  g1[2] = (int)(((width_elems >> 16) & 0xffffu)    // tensor_dim0[31:16]
                | ((height_rows & 0xffffu) << 16));// tensor_dim1[15:0]
  g1[3] = (int)(((height_rows >> 16) & 0xffffu)    // tensor_dim1[31:16]
                | ((width_elems & 0xffffu) << 16));// tile_dim0
  g1[4] = (int)(height_rows & 0xffffu);            // tile_dim1 (tile_dim2=0)
  g1[5] = (int)row_stride_elems;                   // tensor_dim0_stride[31:0]
  g1[6] = 0;                                       // stride[47:32], dim1_stride lo
  g1[7] = 0;
  i32x4 gz = {0, 0, 0, 0};
#if defined(__clang_major__) && __clang_major__ >= 23
  i32x8 gz8 = {0, 0, 0, 0, 0, 0, 0, 0};
  __builtin_amdgcn_tensor_load_to_lds(g0, g1, gz, gz, gz8, 0);
#else
  __builtin_amdgcn_tensor_load_to_lds(g0, g1, gz, gz, 0);
#endif
}
__device__ __forceinline__ unsigned int lds_off(const void* p) {
  return (unsigned int)(uintptr_t)p;  // low 32 bits of flat addr = LDS offset
}
#endif  // USE_TDM

// ---------------------------------------------------------------- converters
__global__ __launch_bounds__(256) void cvt_f16_kernel(const float* __restrict__ in,
                                                      half_t* __restrict__ out,
                                                      int n) {
  int i = (blockIdx.x * 256 + threadIdx.x) * 8;
  if (i >= n) return;
  float4 a = *(const float4*)(in + i);
  float4 b = *(const float4*)(in + i + 4);
  alignas(16) half_t h[8] = {(half_t)a.x, (half_t)a.y, (half_t)a.z, (half_t)a.w,
                             (half_t)b.x, (half_t)b.y, (half_t)b.z, (half_t)b.w};
  *(int4*)(out + i) = *(const int4*)h;
}

// W f32 [K][N] -> Wt f16 [N][K]  (64x64 LDS tile transpose)
__global__ __launch_bounds__(256) void cvt_transpose_kernel(const float* __restrict__ W,
                                                            half_t* __restrict__ Wt) {
  __shared__ half_t tile[64][72];
  int t = threadIdx.x;
  int bk = blockIdx.x * 64, bn = blockIdx.y * 64;
  int tr = t >> 2, tc = (t & 3) * 16;
  const float* src = W + (size_t)(bk + tr) * IN_F + bn + tc;
#pragma unroll
  for (int j = 0; j < 16; ++j) tile[tc + j][tr] = (half_t)src[j];
  __syncthreads();
  half_t* dst = Wt + (size_t)(bn + tr) * IN_F + bk + tc;
  *(int4*)(dst)     = *(const int4*)(&tile[tr][tc]);
  *(int4*)(dst + 8) = *(const int4*)(&tile[tr][tc + 8]);
}

// ---------------------------------------------------------------- GEMM + bias
// C[M x 1024] = A_f16[M x 1024] @ W (as Wt_f16 [n][k]) + bias ; out f16 or f32
#define GTK 32
#define LDSS 40  // 32 + 8 halves pad -> conflict-free b128 fragment loads

template <int OUT_F16>
__global__ __launch_bounds__(256) void gemm_bias_kernel(
    const half_t* __restrict__ A, const half_t* __restrict__ Bt,
    const float* __restrict__ bias, half_t* __restrict__ Ch,
    float* __restrict__ Cf) {
  __shared__ half_t sA[2][128 * LDSS];
  __shared__ half_t sB[2][128 * LDSS];
  const int t = threadIdx.x, lane = t & 31, w = t >> 5;
  const int wm = w >> 1, wn = w & 1;  // 4x2 wave grid; wave tile 32x64
  const int bm = blockIdx.y * 128, bn = blockIdx.x * 128;

  v8f zz = {};
  v8f acc[2][4];
#pragma unroll
  for (int i = 0; i < 2; ++i)
#pragma unroll
    for (int j = 0; j < 4; ++j) acc[i][j] = zz;

  const int KT = IN_F / GTK;  // 32
  int buf = 0;

#if USE_TDM
  const half_t* gA0 = A + (size_t)bm * IN_F;
  const half_t* gB0 = Bt + (size_t)bn * IN_F;
  if (t < 32) {  // wave 0 drives the Tensor Data Mover
    tdm_load_2d(lds_off(&sA[0][0]), gA0, GTK, 128, IN_F, 3, 3);
    tdm_load_2d(lds_off(&sB[0][0]), gB0, GTK, 128, IN_F, 3, 3);
  }
  for (int kt = 0; kt < KT; ++kt) {
    if (kt + 1 < KT) {
      if (t < 32) {
        tdm_load_2d(lds_off(&sA[buf ^ 1][0]), gA0 + (kt + 1) * GTK, GTK, 128,
                    IN_F, 3, 3);
        tdm_load_2d(lds_off(&sB[buf ^ 1][0]), gB0 + (kt + 1) * GTK, GTK, 128,
                    IN_F, 3, 3);
        __builtin_amdgcn_s_wait_tensorcnt(2);  // current buffer's pair done
      }
    } else if (t < 32) {
      __builtin_amdgcn_s_wait_tensorcnt(0);
    }
    __syncthreads();  // current buffer visible to all waves
    const half_t* cA = sA[buf];
    const half_t* cB = sB[buf];
    v16h af0 = ldsA(cA, LDSS, wm * 32, 0, lane);
    v16h af1 = ldsA(cA, LDSS, wm * 32 + 16, 0, lane);
#pragma unroll
    for (int nt = 0; nt < 4; ++nt) {
      v16h bf = ldsB(cB, LDSS, wn * 64 + nt * 16, 0, lane);
      acc[0][nt] = wmma32(af0, bf, acc[0][nt]);
      acc[1][nt] = wmma32(af1, bf, acc[1][nt]);
    }
    __syncthreads();  // reads done before TDM of iter kt+2 rewrites this buffer
    buf ^= 1;
  }
#else
  const int lrow = t >> 1, lpart = (t & 1) * 16;
  const half_t* gA = A + (size_t)(bm + lrow) * IN_F + lpart;
  const half_t* gB = Bt + (size_t)(bn + lrow) * IN_F + lpart;
  const int lo = lrow * LDSS + lpart;
  int4 ra0 = *(const int4*)(gA);
  int4 ra1 = *(const int4*)(gA + 8);
  int4 rb0 = *(const int4*)(gB);
  int4 rb1 = *(const int4*)(gB + 8);
  *(int4*)&sA[0][lo] = ra0;  *(int4*)&sA[0][lo + 8] = ra1;
  *(int4*)&sB[0][lo] = rb0;  *(int4*)&sB[0][lo + 8] = rb1;
  __syncthreads();
  for (int kt = 0; kt < KT; ++kt) {
    if (kt + 1 < KT) {
      const half_t* pa = gA + (kt + 1) * GTK;
      const half_t* pb = gB + (kt + 1) * GTK;
      ra0 = *(const int4*)(pa); ra1 = *(const int4*)(pa + 8);
      rb0 = *(const int4*)(pb); rb1 = *(const int4*)(pb + 8);
    }
    const half_t* cA = sA[buf];
    const half_t* cB = sB[buf];
    v16h af0 = ldsA(cA, LDSS, wm * 32, 0, lane);
    v16h af1 = ldsA(cA, LDSS, wm * 32 + 16, 0, lane);
#pragma unroll
    for (int nt = 0; nt < 4; ++nt) {
      v16h bf = ldsB(cB, LDSS, wn * 64 + nt * 16, 0, lane);
      acc[0][nt] = wmma32(af0, bf, acc[0][nt]);
      acc[1][nt] = wmma32(af1, bf, acc[1][nt]);
    }
    if (kt + 1 < KT) {
      __syncthreads();
      buf ^= 1;
      *(int4*)&sA[buf][lo] = ra0;  *(int4*)&sA[buf][lo + 8] = ra1;
      *(int4*)&sB[buf][lo] = rb0;  *(int4*)&sB[buf][lo + 8] = rb1;
      __syncthreads();
    }
  }
#endif

#pragma unroll
  for (int nt = 0; nt < 4; ++nt) {
    int n = bn + wn * 64 + nt * 16 + (lane & 15);
    float bv = bias[n];
#pragma unroll
    for (int mt = 0; mt < 2; ++mt) {
      int m = bm + wm * 32 + mt * 16 + ((lane >> 4) << 3);
#pragma unroll
      for (int r = 0; r < 8; ++r) {
        float v = acc[mt][nt][r] + bv;
        if (OUT_F16) Ch[(size_t)(m + r) * IN_F + n] = (half_t)v;
        else         Cf[(size_t)(m + r) * IN_F + n] = v;
      }
    }
  }
}

// ---------------------------------------------------------------- flash attention
// Q,K,V,O: f16 [8192][1024] viewed as [b][s][h][64]. Causal, scale 1/8.
__global__ __launch_bounds__(256) void flash_attn_kernel(
    const half_t* __restrict__ Q, const half_t* __restrict__ K,
    const half_t* __restrict__ V, half_t* __restrict__ O) {
  __shared__ half_t sQ[128 * 72];
  __shared__ half_t sK[64 * 72];     // [kpos][d]  (== B-frag layout for Q·Kᵀ)
  __shared__ half_t sVt[64 * 72];    // [d][kpos]  (== B-frag layout for P·V)
  __shared__ half_t sP[8][16 * 72];  // per-wave P staging (C-layout -> A-layout)

  const int t = threadIdx.x, lane = t & 31, w = t >> 5;
  const int b = blockIdx.x >> 4, h = blockIdx.x & 15;
  const int qt = blockIdx.y;
  const int row_base = b * SEQ;
  const int kcol = lane & 15;
  const int q0 = qt * 128 + w * 16 + ((lane >> 4) << 3);  // lane's first q row

#if USE_TDM
  if (t < 32)  // Q tile 128x64 via TDM (waited on with the first K tile)
    tdm_load_2d(lds_off(&sQ[0]),
                Q + (size_t)(row_base + qt * 128) * IN_F + h * HDIM,
                HDIM, 128, IN_F, 4, 3);
#else
  {  // Q tile 128 x 64 -> LDS
    int lr = t >> 1, lp = (t & 1) * 32;
    const half_t* src = Q + (size_t)(row_base + qt * 128 + lr) * IN_F + h * HDIM + lp;
    half_t* dst = &sQ[lr * 72 + lp];
#pragma unroll
    for (int j = 0; j < 4; ++j) *(int4*)(dst + j * 8) = *(const int4*)(src + j * 8);
  }
#endif

  v8f zz = {};
  v8f o[4];
#pragma unroll
  for (int i = 0; i < 4; ++i) o[i] = zz;
  float mrow[8], lsum[8];
#pragma unroll
  for (int r = 0; r < 8; ++r) { mrow[r] = -1e30f; lsum[r] = 0.f; }

  const int nkt = 2 * qt + 2;  // causal: only k-tiles with kt*64 <= q_max
  for (int kt = 0; kt < nkt; ++kt) {
    __syncthreads();  // previous iteration finished with sK/sVt
#if USE_TDM
    if (t < 32)  // K tile 64x64 natural layout via TDM
      tdm_load_2d(lds_off(&sK[0]),
                  K + (size_t)(row_base + kt * 64) * IN_F + h * HDIM,
                  HDIM, 64, IN_F, 4, 3);
#endif
    {  // V tile transposed -> LDS (manual; TDM cannot transpose)
      int lr = t >> 2, lp = (t & 3) * 16;
#if !USE_TDM
      const half_t* srcK = K + (size_t)(row_base + kt * 64 + lr) * IN_F + h * HDIM + lp;
      half_t* dk = &sK[lr * 72 + lp];
      *(int4*)(dk)     = *(const int4*)(srcK);
      *(int4*)(dk + 8) = *(const int4*)(srcK + 8);
#endif
      const half_t* srcV = V + (size_t)(row_base + kt * 64 + lr) * IN_F + h * HDIM + lp;
      alignas(16) half_t tmp[16];
      *(int4*)(tmp)     = *(const int4*)(srcV);
      *(int4*)(tmp + 8) = *(const int4*)(srcV + 8);
#pragma unroll
      for (int j = 0; j < 16; ++j) sVt[(lp + j) * 72 + lr] = tmp[j];
    }
#if USE_TDM
    if (t < 32) __builtin_amdgcn_s_wait_tensorcnt(0);
#endif
    __syncthreads();

    // S = Q Kᵀ
    v16h aq0 = ldsA(sQ, 72, w * 16, 0, lane);
    v16h aq1 = ldsA(sQ, 72, w * 16, 32, lane);
    v8f s[4];
#pragma unroll
    for (int nt = 0; nt < 4; ++nt) {
      v16h bk0 = ldsB(sK, 72, nt * 16, 0, lane);
      v16h bk1 = ldsB(sK, 72, nt * 16, 32, lane);
      v8f c = zz;
      c = wmma32(aq0, bk0, c);
      c = wmma32(aq1, bk1, c);
      s[nt] = c;
    }
    // scale + causal mask
#pragma unroll
    for (int nt = 0; nt < 4; ++nt) {
      int kg = kt * 64 + nt * 16 + kcol;
#pragma unroll
      for (int r = 0; r < 8; ++r) {
        float v = s[nt][r] * 0.125f;
        s[nt][r] = (kg <= q0 + r) ? v : -1e30f;
      }
    }
    // online softmax (row spans 16 lanes within a half-wave)
    float mnew[8], psc[8];
#pragma unroll
    for (int r = 0; r < 8; ++r) {
      float mx = fmaxf(fmaxf(s[0][r], s[1][r]), fmaxf(s[2][r], s[3][r]));
      mx = fmaxf(mx, __shfl_xor(mx, 1, 32));
      mx = fmaxf(mx, __shfl_xor(mx, 2, 32));
      mx = fmaxf(mx, __shfl_xor(mx, 4, 32));
      mx = fmaxf(mx, __shfl_xor(mx, 8, 32));
      mnew[r] = fmaxf(mrow[r], mx);
      psc[r] = __expf(mrow[r] - mnew[r]);
      mrow[r] = mnew[r];
    }
    float rsum[8];
#pragma unroll
    for (int r = 0; r < 8; ++r) rsum[r] = 0.f;
#pragma unroll
    for (int nt = 0; nt < 4; ++nt) {
#pragma unroll
      for (int r = 0; r < 8; ++r) {
        float p = __expf(s[nt][r] - mnew[r]);
        rsum[r] += p;
        sP[w][(((lane >> 4) << 3) + r) * 72 + nt * 16 + kcol] = (half_t)p;
      }
    }
#pragma unroll
    for (int r = 0; r < 8; ++r) {
      float rs = rsum[r];
      rs += __shfl_xor(rs, 1, 32);
      rs += __shfl_xor(rs, 2, 32);
      rs += __shfl_xor(rs, 4, 32);
      rs += __shfl_xor(rs, 8, 32);
      lsum[r] = lsum[r] * psc[r] + rs;
    }
#pragma unroll
    for (int dn = 0; dn < 4; ++dn)
#pragma unroll
      for (int r = 0; r < 8; ++r) o[dn][r] *= psc[r];

    __syncthreads();  // sP visible

    // O += P · V
    v16h ap0 = ldsA(sP[w], 72, 0, 0, lane);
    v16h ap1 = ldsA(sP[w], 72, 0, 32, lane);
#pragma unroll
    for (int dn = 0; dn < 4; ++dn) {
      v16h bv0 = ldsB(sVt, 72, dn * 16, 0, lane);
      v16h bv1 = ldsB(sVt, 72, dn * 16, 32, lane);
      o[dn] = wmma32(ap0, bv0, o[dn]);
      o[dn] = wmma32(ap1, bv1, o[dn]);
    }
  }

#pragma unroll
  for (int dn = 0; dn < 4; ++dn) {
    int d = dn * 16 + kcol;
#pragma unroll
    for (int r = 0; r < 8; ++r) {
      float v = o[dn][r] / lsum[r];
      O[(size_t)(row_base + q0 + r) * IN_F + h * HDIM + d] = (half_t)v;
    }
  }
}

// ---------------------------------------------------------------- launcher
extern "C" void kernel_launch(void* const* d_in, const int* in_sizes, int n_in,
                              void* d_out, int out_size, void* d_ws, size_t ws_size,
                              hipStream_t stream) {
  const float* x  = (const float*)d_in[0];
  const float* wq = (const float*)d_in[1];
  const float* bq = (const float*)d_in[2];
  const float* wk = (const float*)d_in[3];
  const float* bk = (const float*)d_in[4];
  const float* wv = (const float*)d_in[5];
  const float* bv = (const float*)d_in[6];
  const float* wo = (const float*)d_in[7];
  const float* bo = (const float*)d_in[8];
  float* out = (float*)d_out;

  char* ws = (char*)d_ws;
  size_t off = 0;
  const size_t actN = (size_t)ROWS * IN_F;         // 8M elems
  const size_t wN   = (size_t)IN_F * IN_F;         // 1M elems
  half_t* xh  = (half_t*)(ws + off); off += actN * 2;   // reused for attn out
  half_t* wqt = (half_t*)(ws + off); off += wN * 2;
  half_t* wkt = (half_t*)(ws + off); off += wN * 2;
  half_t* wvt = (half_t*)(ws + off); off += wN * 2;
  half_t* wot = (half_t*)(ws + off); off += wN * 2;
  half_t* qh  = (half_t*)(ws + off); off += actN * 2;
  half_t* kh  = (half_t*)(ws + off); off += actN * 2;
  half_t* vh  = (half_t*)(ws + off); off += actN * 2;   // total 72 MB
  half_t* ah  = xh;  // x no longer needed after the QKV GEMMs

  cvt_f16_kernel<<<(int)(actN / 2048), 256, 0, stream>>>(x, xh, (int)actN);
  dim3 tg(IN_F / 64, IN_F / 64);
  cvt_transpose_kernel<<<tg, 256, 0, stream>>>(wq, wqt);
  cvt_transpose_kernel<<<tg, 256, 0, stream>>>(wk, wkt);
  cvt_transpose_kernel<<<tg, 256, 0, stream>>>(wv, wvt);
  cvt_transpose_kernel<<<tg, 256, 0, stream>>>(wo, wot);

  dim3 gg(IN_F / 128, ROWS / 128);  // (8, 64)
  gemm_bias_kernel<1><<<gg, 256, 0, stream>>>(xh, wqt, bq, qh, nullptr);
  gemm_bias_kernel<1><<<gg, 256, 0, stream>>>(xh, wkt, bk, kh, nullptr);
  gemm_bias_kernel<1><<<gg, 256, 0, stream>>>(xh, wvt, bv, vh, nullptr);

  dim3 ag(BATCH * HEADS, SEQ / 128);  // (64, 16)
  flash_attn_kernel<<<ag, 256, 0, stream>>>(qh, kh, vh, ah);

  gemm_bias_kernel<0><<<gg, 256, 0, stream>>>(ah, wot, bo, nullptr, out);
}
